// self_learning_48558900248652
// MI455X (gfx1250) — matrix operand order repaired and checked by reading, not compiled, not run on hardware
//
#include <hip/hip_runtime.h>

// ---------------------------------------------------------------- constants
#define BSZ   32
#define LTOK  256          // tokens per modality
#define DIM   256
#define SEQ   520          // S_MAX
#define MROWS (BSZ*SEQ)    // 16640, multiple of 64
#define HEADS 8
#define HD    32
#define MLPD  1024
#define NEGC  (-1e9f)

static const size_t O_TGT = (size_t)BSZ * 519 * 256;      // out2 offset
static const size_t O_LM  = 2 * O_TGT;                    // out3 offset
static const size_t O_CLS = O_LM + (size_t)BSZ * 519;     // out4 offset

typedef __attribute__((ext_vector_type(16))) _Float16 v16h;
typedef __attribute__((ext_vector_type(8)))  float    v8f;

union FragH { v16h v; _Float16 e[16]; unsigned u[8]; uint4 q[2]; };
union V8    { v8f  v; float e[8]; };

// ---------------------------------------------------------------- helpers
__device__ inline unsigned hash3(unsigned a, unsigned b, unsigned c) {
  unsigned h = a * 0x9E3779B1u ^ b * 0x85EBCA77u ^ c * 0xC2B2AE3Du;
  h ^= h >> 16; h *= 0x7FEB352Du; h ^= h >> 15; h *= 0x846CA68Bu; h ^= h >> 16;
  return h;
}
__device__ inline float u01(unsigned h) { return (h >> 8) * (1.0f / 16777216.0f); }

__device__ inline float gelu_tanh(float x) {
  float x3 = x * x * x;
  return 0.5f * x * (1.0f + tanhf(0.7978845608028654f * (x + 0.044715f * x3)));
}

// ---------------------------------------------------------------- f32 -> f16
__global__ void cvt_f16_kernel(const float* __restrict__ in,
                               _Float16* __restrict__ out, int n) {
  for (int i = blockIdx.x * blockDim.x + threadIdx.x; i < n;
       i += gridDim.x * blockDim.x)
    out[i] = (_Float16)in[i];
}

// ---------------------------------------------------------------- lengths
__global__ void lens_kernel(const float* __restrict__ ma,
                            const float* __restrict__ mb,
                            int* __restrict__ lens) {
  int t = threadIdx.x;                     // 64 threads: 32 for a, 32 for b
  if (t >= 64) return;
  const float* m = (t < 32) ? (ma + t * LTOK) : (mb + (t - 32) * LTOK);
  int c = 0;
  for (int i = 0; i < LTOK; i++) c += (m[i] > 0.f) ? 1 : 0;
  lens[t] = c;
}

// ---------------------------------------------------------------- MLM + pack
__global__ void build_seq_kernel(const float* __restrict__ fa,
                                 const float* __restrict__ fb,
                                 const float* __restrict__ cls,
                                 const float* __restrict__ mfill,
                                 const int* __restrict__ lens,
                                 float* __restrict__ x,
                                 float* __restrict__ out) {
  int blk = blockIdx.x;
  int b = blk / SEQ, s = blk % SEQ, d = threadIdx.x;
  int la = lens[b], lb = lens[32 + b];
  float sv, tv, lm = 0.f;
  if (s == 0) {
    sv = cls[b * DIM + d];
    tv = mfill[d];
  } else if (s <= la + lb) {
    int which, tok; const float* feat;
    if (s <= la) { which = 0; tok = s - 1;      feat = fa; }
    else         { which = 1; tok = s - 1 - la; feat = fb; }
    unsigned h1 = hash3(which,          b, tok);
    unsigned h2 = hash3(which ^ 0x111u, b, tok);
    unsigned h3 = hash3(which ^ 0x222u, b, tok);
    unsigned h4 = hash3(which ^ 0x333u, b, tok);
    bool masked = u01(h1) < 0.5f;
    bool lp = u01(h2) < 0.8f;
    bool rp = (u01(h3) < 0.5f) && !lp;
    float orig = feat[((size_t)(b * LTOK + tok)) * DIM + d];
    tv = orig; lm = masked ? 1.f : 0.f;
    if (masked && lp)      sv = mfill[d];
    else if (masked && rp) sv = feat[((size_t)(h4 % (BSZ * LTOK))) * DIM + d];
    else                   sv = orig;
  } else {                                   // pad slots
    sv = mfill[d]; tv = mfill[d];
  }
  x[((size_t)(b * SEQ + s)) * DIM + d] = sv;
  if (s > 0) {
    out[O_TGT + ((size_t)(b * 519 + s - 1)) * DIM + d] = tv;
    if (d == 0) out[O_LM + (size_t)b * 519 + (s - 1)] = lm;
  }
}

// ---------------------------------------------------------------- layernorm
__global__ void layernorm_kernel(const float* __restrict__ x,
                                 const float* __restrict__ g,
                                 const float* __restrict__ bb,
                                 _Float16* __restrict__ out) {
  int row = blockIdx.x, d = threadIdx.x;
  __shared__ float red[DIM];
  float v = x[(size_t)row * DIM + d];
  red[d] = v; __syncthreads();
  for (int off = 128; off > 0; off >>= 1) {
    if (d < off) red[d] += red[d + off];
    __syncthreads();
  }
  float mean = red[0] * (1.0f / DIM);
  __syncthreads();
  float c = v - mean;
  red[d] = c * c; __syncthreads();
  for (int off = 128; off > 0; off >>= 1) {
    if (d < off) red[d] += red[d + off];
    __syncthreads();
  }
  float var = red[0] * (1.0f / DIM);
  float r = rsqrtf(var + 1e-5f);
  out[(size_t)row * DIM + d] = (_Float16)(c * r * g[d] + bb[d]);
}

// ---------------------------------------------------------------- WMMA GEMM
// C[M,N] = A[M,K](f16) * W[K,N](f16) + bias, fused epilogues.
// Block tile 64x128, 8 waves, each wave 16x64 (4 accumulators).
// Software-pipelined: next K-step's global loads issue before compute.
// MODE 0: outH = f16(acc+bias)
// MODE 1: outH = f16(gelu(acc+bias))
// MODE 2: outF = resid + acc + bias        (f32)
template <int KD, int ND, int MODE>
__global__ __launch_bounds__(256) void gemm16_kernel(
    const _Float16* __restrict__ A, const _Float16* __restrict__ W,
    const float* __restrict__ bias, const float* __restrict__ resid,
    float* __restrict__ outF, _Float16* __restrict__ outH) {
  __shared__ _Float16 As[64][40];    // 64x32 tile, row stride 80B
  __shared__ _Float16 Bs[32][136];   // 32x128 tile, row stride 272B

  const int t    = threadIdx.x;
  const int wave = t >> 5, lane = t & 31;
  const int wr = wave >> 1;           // wave row block 0..3 (16 rows each)
  const int wc = (wave & 1) * 64;     // wave col block 0 or 64
  const int m0 = blockIdx.y * 64;
  const int n0 = blockIdx.x * 128;

  V8 acc[4];
  for (int s = 0; s < 4; s++)
    for (int i = 0; i < 8; i++) acc[s].e[i] = 0.f;

  const int mload = t >> 2, kload = (t & 3) * 8;    // A: 64 rows x 32 cols
  const int kbld  = t >> 3, nload = (t & 7) * 16;   // B: 32 rows x 128 cols
  const int m     = (lane & 15);                    // frag row within 16
  const int kh    = lane >> 4;

  // prologue: first tile into registers
  uint4 aReg  = *(const uint4*)&A[((size_t)(m0 + mload)) * KD + kload];
  uint4 bReg0 = *(const uint4*)&W[(size_t)kbld * ND + n0 + nload];
  uint4 bReg1 = *(const uint4*)&W[(size_t)kbld * ND + n0 + nload + 8];

  for (int k0 = 0; k0 < KD; k0 += 32) {
    __syncthreads();
    *(uint4*)&As[mload][kload]     = aReg;
    *(uint4*)&Bs[kbld][nload]      = bReg0;
    *(uint4*)&Bs[kbld][nload + 8]  = bReg1;
    __syncthreads();

    if (k0 + 32 < KD) {               // next tile: overlap with WMMA compute
      aReg  = *(const uint4*)&A[((size_t)(m0 + mload)) * KD + k0 + 32 + kload];
      bReg0 = *(const uint4*)&W[((size_t)(k0 + 32 + kbld)) * ND + n0 + nload];
      bReg1 = *(const uint4*)&W[((size_t)(k0 + 32 + kbld)) * ND + n0 + nload + 8];
    }

    FragH af;                         // ISA A layout: contiguous b128 reads
    const int mrow = wr * 16 + m;
    af.q[0] = *(const uint4*)&As[mrow][kh * 8];
    af.q[1] = *(const uint4*)&As[mrow][16 + kh * 8];

    for (int s = 0; s < 4; s++) {     // B layout: lane = N col, 16 K elems (table order)
      FragH bf;
      for (int i = 0; i < 16; i++)
        bf.e[i] = Bs[(i < 8) ? (kh * 8 + i) : (16 + kh * 8 + i - 8)][wc + s * 16 + m];
      acc[s].v = __builtin_amdgcn_wmma_f32_16x16x32_f16(
          false, af.v, false, bf.v, (short)0, acc[s].v, false, false);
    }
  }

  const int rbase = m0 + wr * 16 + (lane >> 4) * 8;   // C layout rows
  for (int s = 0; s < 4; s++) {
    const int cg = n0 + wc + s * 16 + (lane & 15);
    const float bi = bias[cg];
    for (int i = 0; i < 8; i++) {
      const size_t r = (size_t)(rbase + i);
      float v = acc[s].e[i] + bi;
      if (MODE == 0)      outH[r * ND + cg] = (_Float16)v;
      else if (MODE == 1) outH[r * ND + cg] = (_Float16)gelu_tanh(v);
      else                outF[r * ND + cg] = resid[r * ND + cg] + v;
    }
  }
}

// ---------------------------------------------------------------- attention
// One wave per (b, head, 16-query tile). Transposed dataflow:
//   S^T[key,q] = K (A-frag, b128 LDS) x Q^T (B-frag, staged once, pre-scaled)
//   O^T[hd,q]  = V^T (A-frag, b128 LDS) x P^T (B-frag, per-element LDS reads)
// Validity is front-contiguous (packed sequence): key valid <=> key < nv,
// so the key loop runs only ceil(nv/32) tiles and only the last tile masks.
// K/V global loads for step kt+1 are register-double-buffered over compute.
__global__ __launch_bounds__(32) void attention_kernel(
    const _Float16* __restrict__ qkv, const int* __restrict__ lens,
    _Float16* __restrict__ attn) {
  const int qt = blockIdx.x, hh = blockIdx.y, b = blockIdx.z;
  const int lane = threadIdx.x;
  __shared__ _Float16 Qt[32][24];   // [hd][query]  stride 48B
  __shared__ _Float16 Ks[32][40];   // [key][hd]    stride 80B
  __shared__ _Float16 Vt[32][40];   // [hd][key]    stride 80B
  __shared__ _Float16 Pt[32][24];   // [key][query] stride 48B

  const uint4 z4 = make_uint4(0u, 0u, 0u, 0u);
  const int m = lane & 15, kh = lane >> 4;
  const int col = lane & 15, rg = lane >> 4;

  const int nv  = 1 + lens[b] + lens[32 + b];   // valid prefix length (>= 129)
  const int ktn = (nv + 31) >> 5;               // key tiles actually needed

  // stage Q transposed (once), pre-scaled by 1/sqrt(hd)
  if (lane < 16) {
    int q = qt * 16 + lane;
    FragH qa, qb2;
    if (q < SEQ) {
      const uint4* src =
          (const uint4*)&qkv[((size_t)(b * SEQ + q)) * (3 * DIM) + hh * HD];
      qa.q[0] = src[0]; qa.q[1] = src[1]; qb2.q[0] = src[2]; qb2.q[1] = src[3];
    } else {
      qa.q[0] = z4; qa.q[1] = z4; qb2.q[0] = z4; qb2.q[1] = z4;
    }
    const _Float16 hs = (_Float16)0.17677669529663687f;
    for (int k = 0; k < 16; k++) {
      Qt[k][lane]      = qa.e[k] * hs;
      Qt[16 + k][lane] = qb2.e[k] * hs;
    }
  }
  __syncthreads();
  FragH qb;                          // B-frag Q^T: lane = query col, 16 hd elems
  for (int i = 0; i < 16; i++)
    qb.e[i] = Qt[(i < 8) ? (kh * 8 + i) : (16 + kh * 8 + i - 8)][m];

  V8 ot0, ot1;                       // O^T: hd rows [0..15], [16..31] x queries
  for (int i = 0; i < 8; i++) { ot0.e[i] = 0.f; ot1.e[i] = 0.f; }
  float mq = -1e30f, sq = 0.f;       // per-query running max / sum

  // prologue: kt=0 K/V rows into registers (key = lane < nv always valid)
  FragH kra, krb, vra, vrb;
  {
    const size_t base = ((size_t)(b * SEQ + lane)) * (3 * DIM);
    const uint4* sk = (const uint4*)&qkv[base + DIM + hh * HD];
    const uint4* sv = (const uint4*)&qkv[base + 2 * DIM + hh * HD];
    kra.q[0] = sk[0]; kra.q[1] = sk[1]; krb.q[0] = sk[2]; krb.q[1] = sk[3];
    vra.q[0] = sv[0]; vra.q[1] = sv[1]; vrb.q[0] = sv[2]; vrb.q[1] = sv[3];
  }

  for (int kt = 0; kt < ktn; kt++) {
    const int kbase = kt * 32;
    __syncthreads();
    {                                 // regs -> LDS (K normal, V transposed)
      uint4* dk = (uint4*)&Ks[lane][0];
      dk[0] = kra.q[0]; dk[1] = kra.q[1]; dk[2] = krb.q[0]; dk[3] = krb.q[1];
      for (int k = 0; k < 16; k++) {
        Vt[k][lane]      = vra.e[k];
        Vt[16 + k][lane] = vrb.e[k];
      }
    }
    __syncthreads();

    if (kt + 1 < ktn) {               // next K/V tile: overlap with compute
      const int key = (kt + 1) * 32 + lane;
      if (key < SEQ) {
        const size_t base = ((size_t)(b * SEQ + key)) * (3 * DIM);
        const uint4* sk = (const uint4*)&qkv[base + DIM + hh * HD];
        const uint4* sv = (const uint4*)&qkv[base + 2 * DIM + hh * HD];
        kra.q[0] = sk[0]; kra.q[1] = sk[1]; krb.q[0] = sk[2]; krb.q[1] = sk[3];
        vra.q[0] = sv[0]; vra.q[1] = sv[1]; vrb.q[0] = sv[2]; vrb.q[1] = sv[3];
      } else {
        kra.q[0] = z4; kra.q[1] = z4; krb.q[0] = z4; krb.q[1] = z4;
        vra.q[0] = z4; vra.q[1] = z4; vrb.q[0] = z4; vrb.q[1] = z4;
      }
    }

    FragH ka0, ka1;                  // K A-frags: keys 0-15 / 16-31, b128 LDS
    ka0.q[0] = *(const uint4*)&Ks[m][kh * 8];
    ka0.q[1] = *(const uint4*)&Ks[m][16 + kh * 8];
    ka1.q[0] = *(const uint4*)&Ks[16 + m][kh * 8];
    ka1.q[1] = *(const uint4*)&Ks[16 + m][16 + kh * 8];

    v8f zc = {0.f, 0.f, 0.f, 0.f, 0.f, 0.f, 0.f, 0.f};
    V8 s0, s1;                       // S^T tiles: rows = keys, cols = queries
    s0.v = __builtin_amdgcn_wmma_f32_16x16x32_f16(
        false, ka0.v, false, qb.v, (short)0, zc, false, false);
    s1.v = __builtin_amdgcn_wmma_f32_16x16x32_f16(
        false, ka1.v, false, qb.v, (short)0, zc, false, false);

    if (kbase + 32 > nv) {           // uniform branch: only last tile masks
      const int rem = nv - kbase - rg * 8;        // valid elems in sub0 group
      for (int i = 0; i < 8; i++) {
        s0.e[i] = (i < rem)      ? s0.e[i] : NEGC;
        s1.e[i] = (i < rem - 16) ? s1.e[i] : NEGC;
      }
    }
    float mx = -1e30f;
    for (int i = 0; i < 8; i++) mx = fmaxf(mx, fmaxf(s0.e[i], s1.e[i]));
    mx = fmaxf(mx, __shfl_xor(mx, 16, 32));       // combine key halves
    float nm = fmaxf(mq, mx);
    float al = __expf(mq - nm);
    mq = nm;
    float ps = 0.f;
    for (int i = 0; i < 8; i++) {
      float p0 = __expf(s0.e[i] - nm);
      float p1 = __expf(s1.e[i] - nm);
      ps += p0 + p1;
      Pt[rg * 8 + i][col]      = (_Float16)p0;
      Pt[16 + rg * 8 + i][col] = (_Float16)p1;
    }
    ps += __shfl_xor(ps, 16, 32);
    sq = sq * al + ps;
    for (int i = 0; i < 8; i++) { ot0.e[i] *= al; ot1.e[i] *= al; }
    __syncthreads();

    FragH pb;                        // P^T B-frag: lane = query col, 16 key elems
    for (int i = 0; i < 16; i++)
      pb.e[i] = Pt[(i < 8) ? (kh * 8 + i) : (16 + kh * 8 + i - 8)][col];
    FragH va0, va1;                  // V^T A-frags: hd 0-15 / 16-31, b128 LDS
    va0.q[0] = *(const uint4*)&Vt[m][kh * 8];
    va0.q[1] = *(const uint4*)&Vt[m][16 + kh * 8];
    va1.q[0] = *(const uint4*)&Vt[16 + m][kh * 8];
    va1.q[1] = *(const uint4*)&Vt[16 + m][16 + kh * 8];

    ot0.v = __builtin_amdgcn_wmma_f32_16x16x32_f16(
        false, va0.v, false, pb.v, (short)0, ot0.v, false, false);
    ot1.v = __builtin_amdgcn_wmma_f32_16x16x32_f16(
        false, va1.v, false, pb.v, (short)0, ot1.v, false, false);
  }

  const int q = qt * 16 + col;
  if (q < SEQ) {
    const float inv = 1.0f / sq;
    const size_t base = ((size_t)(b * SEQ + q)) * DIM + hh * HD;
    for (int i = 0; i < 8; i++) {
      attn[base + rg * 8 + i]      = (_Float16)(ot0.e[i] * inv);
      attn[base + 16 + rg * 8 + i] = (_Float16)(ot1.e[i] * inv);
    }
  }
}

// ---------------------------------------------------------------- scatter out
__global__ void write_out_kernel(const float* __restrict__ x,
                                 float* __restrict__ out) {
  for (size_t i = (size_t)blockIdx.x * blockDim.x + threadIdx.x;
       i < (size_t)MROWS * DIM; i += (size_t)gridDim.x * blockDim.x) {
    int d = (int)(i & 255);
    int row = (int)(i >> 8);
    int b = row / SEQ, s = row % SEQ;
    float v = x[i];
    if (s == 0) out[O_CLS + (size_t)b * DIM + d] = v;
    else        out[((size_t)(b * 519 + s - 1)) * DIM + d] = v;
  }
}

// ---------------------------------------------------------------- launch
extern "C" void kernel_launch(void* const* d_in, const int* in_sizes, int n_in,
                              void* d_out, int out_size, void* d_ws,
                              size_t ws_size, hipStream_t stream) {
  (void)in_sizes; (void)n_in; (void)out_size; (void)ws_size;
  const float* fa    = (const float*)d_in[0];
  const float* fb    = (const float*)d_in[1];
  const float* ma    = (const float*)d_in[2];
  const float* mb    = (const float*)d_in[3];
  const float* cls   = (const float*)d_in[4];
  const float* mfill = (const float*)d_in[5];
  const float* wqkv  = (const float*)d_in[6];
  const float* bqkv  = (const float*)d_in[7];
  const float* wo    = (const float*)d_in[8];
  const float* bo    = (const float*)d_in[9];
  const float* w1    = (const float*)d_in[10];
  const float* b1    = (const float*)d_in[11];
  const float* w2    = (const float*)d_in[12];
  const float* b2    = (const float*)d_in[13];
  const float* ln1g  = (const float*)d_in[14];
  const float* ln1b  = (const float*)d_in[15];
  const float* ln2g  = (const float*)d_in[16];
  const float* ln2b  = (const float*)d_in[17];
  float* out = (float*)d_out;

  char* ws = (char*)d_ws;
  size_t off = 0;
  auto alloc = [&](size_t n) {
    size_t o = off; off = (off + n + 255) & ~(size_t)255; return o;
  };
  float*     x      = (float*)(ws + alloc((size_t)MROWS * DIM * 4));
  _Float16*  h16    = (_Float16*)(ws + alloc((size_t)MROWS * DIM * 2));
  _Float16*  qkv16  = (_Float16*)(ws + alloc((size_t)MROWS * 3 * DIM * 2));
  _Float16*  attn16 = (_Float16*)(ws + alloc((size_t)MROWS * DIM * 2));
  _Float16*  mlp16  = (_Float16*)(ws + alloc((size_t)MROWS * MLPD * 2));
  _Float16*  wqkv16 = (_Float16*)(ws + alloc((size_t)4 * DIM * 3 * DIM * 2));
  _Float16*  wo16   = (_Float16*)(ws + alloc((size_t)4 * DIM * DIM * 2));
  _Float16*  w116   = (_Float16*)(ws + alloc((size_t)4 * DIM * MLPD * 2));
  _Float16*  w216   = (_Float16*)(ws + alloc((size_t)4 * MLPD * DIM * 2));
  int*       lens   = (int*)(ws + alloc(256));

  cvt_f16_kernel<<<512, 256, 0, stream>>>(wqkv, wqkv16, 4 * DIM * 3 * DIM);
  cvt_f16_kernel<<<256, 256, 0, stream>>>(wo,   wo16,   4 * DIM * DIM);
  cvt_f16_kernel<<<512, 256, 0, stream>>>(w1,   w116,   4 * DIM * MLPD);
  cvt_f16_kernel<<<512, 256, 0, stream>>>(w2,   w216,   4 * MLPD * DIM);
  lens_kernel<<<1, 64, 0, stream>>>(ma, mb, lens);
  build_seq_kernel<<<BSZ * SEQ, DIM, 0, stream>>>(fa, fb, cls, mfill, lens, x,
                                                  out);
  for (int l = 0; l < 4; l++) {
    layernorm_kernel<<<MROWS, DIM, 0, stream>>>(x, ln1g + l * DIM,
                                                ln1b + l * DIM, h16);
    gemm16_kernel<DIM, 3 * DIM, 0><<<dim3(6, MROWS / 64), 256, 0, stream>>>(
        h16, wqkv16 + (size_t)l * DIM * 3 * DIM, bqkv + l * 3 * DIM, nullptr,
        nullptr, qkv16);
    attention_kernel<<<dim3(33, HEADS, BSZ), 32, 0, stream>>>(qkv16, lens,
                                                              attn16);
    gemm16_kernel<DIM, DIM, 2><<<dim3(2, MROWS / 64), 256, 0, stream>>>(
        attn16, wo16 + (size_t)l * DIM * DIM, bo + l * DIM, x, x, nullptr);
    layernorm_kernel<<<MROWS, DIM, 0, stream>>>(x, ln2g + l * DIM,
                                                ln2b + l * DIM, h16);
    gemm16_kernel<DIM, MLPD, 1><<<dim3(8, MROWS / 64), 256, 0, stream>>>(
        h16, w116 + (size_t)l * DIM * MLPD, b1 + l * MLPD, nullptr, nullptr,
        mlp16);
    gemm16_kernel<MLPD, DIM, 2><<<dim3(2, MROWS / 64), 256, 0, stream>>>(
        mlp16, w216 + (size_t)l * MLPD * DIM, b2 + l * DIM, x, x, nullptr);
  }
  write_out_kernel<<<1024, 256, 0, stream>>>(x, out);
}
